// ManualAttentionRNN_1451698946233
// MI455X (gfx1250) — compile-verified, matrix operand order checked by reading, not run on hardware
//
#include <hip/hip_runtime.h>
#include <cstdint>

// ---------------------------------------------------------------------------
// Problem constants (from reference)
// ---------------------------------------------------------------------------
#define BB 8
#define TT 256
#define EE 512
#define HH 1024
#define VV 32000
#define BT (BB * TT)          // 2048
#define NEG_INF (-1e30f)

typedef __attribute__((ext_vector_type(16))) __bf16 v16bf;
typedef __attribute__((ext_vector_type(8)))  __bf16 v8bf;
typedef __attribute__((ext_vector_type(8)))  float  v8f;

// ---------------------------------------------------------------------------
// WMMA helpers (CDNA5 16x16x32 bf16, f32 accumulate)
// ISA 7.12.2: 16-bit A (16x32) fragment, lane L: M = L&15, and the 16 bf16
// values are two contiguous 8-element chunks at K = (L>>4)*8 and 16+(L>>4)*8.
// B = W^T with W[N,K] row-major has the same per-lane layout over W rows.
// ---------------------------------------------------------------------------
__device__ __forceinline__ v16bf load_frag_k(const __bf16* __restrict__ row,
                                             int k0, int laneHi) {
  const v8bf lo = *(const v8bf*)(row + k0 + laneHi * 8);
  const v8bf hi = *(const v8bf*)(row + k0 + 16 + laneHi * 8);
  return __builtin_shufflevector(lo, hi, 0, 1, 2, 3, 4, 5, 6, 7,
                                 8, 9, 10, 11, 12, 13, 14, 15);
}

__device__ __forceinline__ v8f wmma_bf16(v16bf a, v16bf b, v8f c) {
  // (neg_a, A, neg_b, B, c_mod, C, reuse_a, reuse_b)
  return __builtin_amdgcn_wmma_f32_16x16x32_bf16(false, a, false, b,
                                                 (short)0, c, false, false);
}

// Async DMA: global -> LDS, 8 bytes per lane, tracked by ASYNCcnt.
// VDST operand = per-lane LDS byte offset (low 32 bits of a generic LDS ptr).
__device__ __forceinline__ void async_g2l_b64(unsigned lds_off, const void* g) {
  asm volatile("global_load_async_to_lds_b64 %0, %1, off"
               :: "v"(lds_off), "v"(g) : "memory");
}
__device__ __forceinline__ void wait_async0() {
  asm volatile("s_wait_asynccnt 0" ::: "memory");
}

// ---------------------------------------------------------------------------
// Generic bf16 WMMA GEMM:  C[M,N](f32) = A[M,K](bf16) x B (+ bias[N]) (+ C)
//   BTRANS=true : B operand is W[N,K] row-major (C = A @ W^T) — contiguous
//   BTRANS=false: B operand is B[K,N] row-major (C = A @ B)   — gathered
// Block: 256 threads = 8 waves; wave tile = 16(M) x 32(N); block = 32 x 128.
// The block's 32x32 A tile is staged into LDS with double-buffered
// global_load_async_to_lds_b64 (ASYNCcnt), shared by the 4 waves per M tile.
// Requires M%32==0, N%128==0, K%32==0 (true for all call sites here).
// ---------------------------------------------------------------------------
template <bool BTRANS>
__global__ void __launch_bounds__(256)
gemm_wmma_bf16(const __bf16* __restrict__ A, const __bf16* __restrict__ B,
               float* __restrict__ C, const float* __restrict__ bias,
               int M, int N, int K, int lda, int ldb, int ldc,
               long strideA, long strideB, long strideC, int accumulate) {
  __shared__ __bf16 tileA[2][32 * 32];   // 2 x 2KB double buffer

  const int tid    = threadIdx.x;
  const int lane   = tid & 31;
  const int wave   = tid >> 5;
  const int laneM  = lane & 15;
  const int laneHi = lane >> 4;

  const int mloc = (wave >> 2) * 16;                   // 0 or 16 within block
  const int m0   = blockIdx.y * 32 + mloc;
  const int n0   = blockIdx.x * 128 + (wave & 3) * 32; // two 16-wide N tiles
  const int bz   = blockIdx.z;

  A += bz * strideA;
  B += bz * strideB;
  C += bz * strideC;

  const __bf16* Ablk = A + (long)blockIdx.y * 32 * lda;  // block's 32 A rows
  const int col0 = n0 + laneM;
  const int col1 = n0 + 16 + laneM;
  const __bf16* Brow0 = B + (long)col0 * ldb;  // BTRANS row pointers
  const __bf16* Brow1 = B + (long)col1 * ldb;

  // staging: 256 threads x 8B = 2KB = 32 rows x 32 bf16
  const int srow = tid >> 3;      // 0..31
  const int scol = (tid & 7) * 4; // 0,4,...,28 (4 bf16 = 8B)

  // prologue: stage k-chunk 0 into buffer 0
  async_g2l_b64((unsigned)(uintptr_t)&tileA[0][srow * 32 + scol],
                Ablk + (long)srow * lda + scol);
  wait_async0();
  __syncthreads();

  v8f c0, c1;
  if (accumulate) {
#pragma unroll
    for (int r = 0; r < 8; ++r) {
      const long row = m0 + laneHi * 8 + r;
      c0[r] = C[row * (long)ldc + col0];
      c1[r] = C[row * (long)ldc + col1];
    }
  } else {
    c0 = {};
    c1 = {};
  }

  for (int k0 = 0; k0 < K; k0 += 32) {
    const int buf = (k0 >> 5) & 1;
    if (k0 + 32 < K) {  // stage next chunk while computing this one
      async_g2l_b64((unsigned)(uintptr_t)&tileA[buf ^ 1][srow * 32 + scol],
                    Ablk + (long)srow * lda + (k0 + 32) + scol);
    }

    v16bf a = load_frag_k(&tileA[buf][(mloc + laneM) * 32], 0, laneHi);
    v16bf b0, b1;
    if (BTRANS) {
      __builtin_prefetch(Brow0 + k0 + 128, 0, 1);  // -> global_prefetch_b8
      b0 = load_frag_k(Brow0, k0, laneHi);
      b1 = load_frag_k(Brow1, k0, laneHi);
    } else {
      const int kb = k0 + laneHi * 8;
#pragma unroll
      for (int i = 0; i < 8; ++i) {
        b0[i]     = B[(long)(kb + i) * ldb + col0];
        b0[i + 8] = B[(long)(kb + 16 + i) * ldb + col0];
        b1[i]     = B[(long)(kb + i) * ldb + col1];
        b1[i + 8] = B[(long)(kb + 16 + i) * ldb + col1];
      }
    }
    c0 = wmma_bf16(a, b0, c0);
    c1 = wmma_bf16(a, b1, c1);

    wait_async0();      // next buffer landed in LDS
    __syncthreads();    // all waves done reading current buffer
  }

  const float bv0 = bias ? bias[col0] : 0.f;
  const float bv1 = bias ? bias[col1] : 0.f;
#pragma unroll
  for (int r = 0; r < 8; ++r) {
    const long row = m0 + laneHi * 8 + r;
    C[row * (long)ldc + col0] = c0[r] + bv0;
    C[row * (long)ldc + col1] = c1[r] + bv1;
  }
}

// ---------------------------------------------------------------------------
// Persistent recurrent layer: h_t = tanh(xp[:,t,:] + h_{t-1} @ Whh^T)
// One workgroup, 1024 threads = 32 waves. h lives in LDS as bf16 (16 rows,
// rows 8..15 stay zero; only rows 0..7 = batch are valid). Each wave owns two
// N tiles (wave, wave+32) of 16 columns; per step: 2 x (K=1024/32) WMMAs.
// The asm memory barrier in the t-loop stops LLVM from hoisting the (loop-
// invariant) Whh fragment loads out of the loop and spilling them to scratch;
// Whh (2MB bf16) stays L2-resident instead.
// ---------------------------------------------------------------------------
__global__ void __launch_bounds__(1024)
rnn_layer_wmma(const float* __restrict__ xp, const __bf16* __restrict__ Whh,
               float* __restrict__ outF, __bf16* __restrict__ outB) {
  __shared__ __bf16 hbuf[16 * HH];  // 32 KB

  const int tid    = threadIdx.x;
  const int lane   = tid & 31;
  const int wave   = tid >> 5;
  const int laneM  = lane & 15;
  const int laneHi = lane >> 4;

  for (int i = tid; i < 16 * HH; i += 1024) hbuf[i] = (__bf16)0.f;
  __syncthreads();

  const int n0a = wave * 16;
  const int n0b = (wave + 32) * 16;
  const __bf16* hrow = &hbuf[laneM * HH];
  const __bf16* Wr0  = Whh + (long)(n0a + laneM) * HH;
  const __bf16* Wr1  = Whh + (long)(n0b + laneM) * HH;

  for (int t = 0; t < TT; ++t) {
    asm volatile("" ::: "memory");  // defeat LICM of Whh loads (no spills)
    v8f c0 = {}, c1 = {};
    for (int k0 = 0; k0 < HH; k0 += 32) {
      v16bf a  = load_frag_k(hrow, k0, laneHi);
      v16bf b0 = load_frag_k(Wr0, k0, laneHi);
      v16bf b1 = load_frag_k(Wr1, k0, laneHi);
      c0 = wmma_bf16(a, b0, c0);
      c1 = wmma_bf16(a, b1, c1);
    }
    __syncthreads();  // all LDS reads of h_{t-1} done before overwrite
    if (laneHi == 0) {
#pragma unroll
      for (int r = 0; r < 8; ++r) {       // r = batch index (C rows 0..7)
        const long base = ((long)r * TT + t) * HH;
        const int ca = n0a + laneM, cb = n0b + laneM;
        const float h0 = tanhf(c0[r] + xp[base + ca]);
        const float h1 = tanhf(c1[r] + xp[base + cb]);
        outF[base + ca] = h0;  outB[base + ca] = (__bf16)h0;
        outF[base + cb] = h1;  outB[base + cb] = (__bf16)h1;
        hbuf[r * HH + ca] = (__bf16)h0;
        hbuf[r * HH + cb] = (__bf16)h1;
      }
    }
    __syncthreads();
  }
}

// ---------------------------------------------------------------------------
// Causal + length masked softmax over scores[b,t,:] (row length 256).
// One block (256 threads = 8 wave32) per row. Writes f32 weights to the
// output tensor and bf16 weights to workspace for the context GEMM.
// ---------------------------------------------------------------------------
__global__ void __launch_bounds__(256)
masked_softmax(const float* __restrict__ scores, const int* __restrict__ lengths,
               float* __restrict__ wOut, __bf16* __restrict__ wB) {
  const int row = blockIdx.x;          // b*T + t
  const int b = row >> 8, t = row & 255;
  const int s = threadIdx.x;
  const int lane = s & 31, wv = s >> 5;
  __shared__ float red[8];
  __shared__ float bcast;

  const float v = scores[(long)row * TT + s];
  const bool valid = (s <= t) && (s < lengths[b]);
  float x = valid ? v : NEG_INF;

  float m = x;
#pragma unroll
  for (int o = 16; o > 0; o >>= 1) m = fmaxf(m, __shfl_xor(m, o, 32));
  if (lane == 0) red[wv] = m;
  __syncthreads();
  if (s == 0) {
    float mm = red[0];
#pragma unroll
    for (int i = 1; i < 8; ++i) mm = fmaxf(mm, red[i]);
    bcast = mm;
  }
  __syncthreads();
  const float mx = bcast;

  const float e = __expf(x - mx);
  float sum = e;
#pragma unroll
  for (int o = 16; o > 0; o >>= 1) sum += __shfl_xor(sum, o, 32);
  __syncthreads();
  if (lane == 0) red[wv] = sum;
  __syncthreads();
  if (s == 0) {
    float ss = 0.f;
#pragma unroll
    for (int i = 0; i < 8; ++i) ss += red[i];
    bcast = ss;
  }
  __syncthreads();

  const float w = e / bcast;
  wOut[(long)row * TT + s] = w;
  wB[(long)row * TT + s] = (__bf16)w;
}

// ---------------------------------------------------------------------------
// Small elementwise kernels
// ---------------------------------------------------------------------------
__global__ void cvt_f32_bf16(const float* __restrict__ in, __bf16* __restrict__ out, long n) {
  long i = (long)blockIdx.x * blockDim.x + threadIdx.x;
  if (i < n) out[i] = (__bf16)in[i];
}

__global__ void tanh_f32_bf16(const float* __restrict__ in, __bf16* __restrict__ out, long n) {
  long i = (long)blockIdx.x * blockDim.x + threadIdx.x;
  if (i < n) out[i] = (__bf16)tanhf(in[i]);
}

__global__ void embed_gather_bf16(const long long* __restrict__ ids,
                                  const float* __restrict__ emb,
                                  __bf16* __restrict__ out) {
  long i = (long)blockIdx.x * blockDim.x + threadIdx.x;  // over BT*E
  if (i >= (long)BT * EE) return;
  const long row = i >> 9;          // /EE
  const int  e   = (int)(i & (EE - 1));
  const long id  = ids[row];
  out[i] = (__bf16)emb[id * EE + e];
}

// ---------------------------------------------------------------------------
// Host-side orchestration
// ---------------------------------------------------------------------------
extern "C" void kernel_launch(void* const* d_in, const int* in_sizes, int n_in,
                              void* d_out, int out_size, void* d_ws, size_t ws_size,
                              hipStream_t stream) {
  (void)in_sizes; (void)n_in; (void)out_size; (void)ws_size;

  const long long* input_ids = (const long long*)d_in[0];
  const int*   lengths    = (const int*)d_in[1];
  const float* embedding  = (const float*)d_in[2];
  const float* wxh_enc0   = (const float*)d_in[3];
  const float* whh_enc0   = (const float*)d_in[4];
  const float* bh_enc0    = (const float*)d_in[5];
  const float* wxh_enc1   = (const float*)d_in[6];
  const float* whh_enc1   = (const float*)d_in[7];
  const float* bh_enc1    = (const float*)d_in[8];
  const float* wxh_dec0   = (const float*)d_in[9];
  const float* whh_dec0   = (const float*)d_in[10];
  const float* bh_dec0    = (const float*)d_in[11];
  const float* wxh_dec1   = (const float*)d_in[12];
  const float* whh_dec1   = (const float*)d_in[13];
  const float* bh_dec1    = (const float*)d_in[14];
  const float* w_att      = (const float*)d_in[15];
  const float* w_cmb_h    = (const float*)d_in[16];
  const float* w_cmb_c    = (const float*)d_in[17];
  const float* b_combine  = (const float*)d_in[18];
  const float* w_out      = (const float*)d_in[19];
  const float* b_out      = (const float*)d_in[20];

  float* logits  = (float*)d_out;                          // [B,T,V]
  float* attnOut = (float*)d_out + (long)BT * VV;          // [B,T,T]

  // -- workspace carve-out ------------------------------------------------
  char* ws = (char*)d_ws;
  size_t off = 0;
  auto alloc = [&](size_t bytes) -> void* {
    void* p = ws + off;
    off += (bytes + 255) & ~(size_t)255;
    return p;
  };
  __bf16* emb_b   = (__bf16*)alloc((size_t)BT * EE * 2);
  __bf16* wxe0_b  = (__bf16*)alloc((size_t)HH * EE * 2);
  __bf16* whe0_b  = (__bf16*)alloc((size_t)HH * HH * 2);
  __bf16* wxe1_b  = (__bf16*)alloc((size_t)HH * HH * 2);
  __bf16* whe1_b  = (__bf16*)alloc((size_t)HH * HH * 2);
  __bf16* wxd0_b  = (__bf16*)alloc((size_t)HH * EE * 2);
  __bf16* whd0_b  = (__bf16*)alloc((size_t)HH * HH * 2);
  __bf16* wxd1_b  = (__bf16*)alloc((size_t)HH * HH * 2);
  __bf16* whd1_b  = (__bf16*)alloc((size_t)HH * HH * 2);
  __bf16* watt_b  = (__bf16*)alloc((size_t)HH * HH * 2);
  __bf16* wch_b   = (__bf16*)alloc((size_t)HH * HH * 2);
  __bf16* wcc_b   = (__bf16*)alloc((size_t)HH * HH * 2);
  __bf16* wout_b  = (__bf16*)alloc((size_t)VV * HH * 2);
  float*  xp_f    = (float*) alloc((size_t)BT * HH * 4);   // reused: xp / transformed / combined
  float*  state_f = (float*) alloc((size_t)BT * HH * 4);   // reused: rnn f32 dump / context
  __bf16* enc0_b  = (__bf16*)alloc((size_t)BT * HH * 2);
  __bf16* enc_b   = (__bf16*)alloc((size_t)BT * HH * 2);
  __bf16* dec0_b  = (__bf16*)alloc((size_t)BT * HH * 2);
  __bf16* dec_b   = (__bf16*)alloc((size_t)BT * HH * 2);
  __bf16* trans_b = (__bf16*)alloc((size_t)BT * HH * 2);
  float*  scor_f  = (float*) alloc((size_t)BT * TT * 4);
  __bf16* attw_b  = (__bf16*)alloc((size_t)BT * TT * 2);
  __bf16* ctx_b   = (__bf16*)alloc((size_t)BT * HH * 2);
  __bf16* comb_b  = (__bf16*)alloc((size_t)BT * HH * 2);

  auto cvt = [&](const float* src, __bf16* dst, long n) {
    cvt_f32_bf16<<<dim3((unsigned)((n + 255) / 256)), dim3(256), 0, stream>>>(src, dst, n);
  };
  auto gemm = [&](bool btrans, const __bf16* A, int lda, const __bf16* B, int ldb,
                  float* C, int ldc, const float* bias, int M, int N, int K,
                  int batch, long sA, long sB, long sC, int acc) {
    dim3 g((unsigned)(N / 128), (unsigned)(M / 32), (unsigned)batch);
    if (btrans)
      gemm_wmma_bf16<true><<<g, 256, 0, stream>>>(A, B, C, bias, M, N, K, lda, ldb, ldc, sA, sB, sC, acc);
    else
      gemm_wmma_bf16<false><<<g, 256, 0, stream>>>(A, B, C, bias, M, N, K, lda, ldb, ldc, sA, sB, sC, acc);
  };

  // -- 1. weight conversions to bf16 -------------------------------------
  cvt(wxh_enc0, wxe0_b, (long)HH * EE);
  cvt(whh_enc0, whe0_b, (long)HH * HH);
  cvt(wxh_enc1, wxe1_b, (long)HH * HH);
  cvt(whh_enc1, whe1_b, (long)HH * HH);
  cvt(wxh_dec0, wxd0_b, (long)HH * EE);
  cvt(whh_dec0, whd0_b, (long)HH * HH);
  cvt(wxh_dec1, wxd1_b, (long)HH * HH);
  cvt(whh_dec1, whd1_b, (long)HH * HH);
  cvt(w_att,    watt_b, (long)HH * HH);
  cvt(w_cmb_h,  wch_b,  (long)HH * HH);
  cvt(w_cmb_c,  wcc_b,  (long)HH * HH);
  cvt(w_out,    wout_b, (long)VV * HH);

  // -- 2. embedding gather ------------------------------------------------
  embed_gather_bf16<<<dim3((BT * EE + 255) / 256), dim3(256), 0, stream>>>(
      input_ids, embedding, emb_b);

  // -- 3. encoder ---------------------------------------------------------
  gemm(true, emb_b, EE, wxe0_b, EE, xp_f, HH, bh_enc0, BT, HH, EE, 1, 0, 0, 0, 0);
  rnn_layer_wmma<<<1, 1024, 0, stream>>>(xp_f, whe0_b, state_f, enc0_b);
  gemm(true, enc0_b, HH, wxe1_b, HH, xp_f, HH, bh_enc1, BT, HH, HH, 1, 0, 0, 0, 0);
  rnn_layer_wmma<<<1, 1024, 0, stream>>>(xp_f, whe1_b, state_f, enc_b);

  // -- 4. decoder (teacher forcing) ---------------------------------------
  gemm(true, emb_b, EE, wxd0_b, EE, xp_f, HH, bh_dec0, BT, HH, EE, 1, 0, 0, 0, 0);
  rnn_layer_wmma<<<1, 1024, 0, stream>>>(xp_f, whd0_b, state_f, dec0_b);
  gemm(true, dec0_b, HH, wxd1_b, HH, xp_f, HH, bh_dec1, BT, HH, HH, 1, 0, 0, 0, 0);
  rnn_layer_wmma<<<1, 1024, 0, stream>>>(xp_f, whd1_b, state_f, dec_b);

  // -- 5. attention -------------------------------------------------------
  // transformed = dec @ w_att  (B normal: B[k=h, n=k'] = w_att[h, k'])
  gemm(false, dec_b, HH, watt_b, HH, xp_f, HH, nullptr, BT, HH, HH, 1, 0, 0, 0, 0);
  cvt(xp_f, trans_b, (long)BT * HH);
  // scores[b] = transformed[b] @ enc[b]^T   (batched, W = enc rows)
  gemm(true, trans_b, HH, enc_b, HH, scor_f, TT, nullptr, TT, TT, HH, BB,
       (long)TT * HH, (long)TT * HH, (long)TT * TT, 0);
  masked_softmax<<<dim3(BT), dim3(TT), 0, stream>>>(scor_f, lengths, attnOut, attw_b);
  // context[b] = weights[b] @ enc[b]        (B normal)
  gemm(false, attw_b, TT, enc_b, HH, state_f, HH, nullptr, TT, HH, TT, BB,
       (long)TT * TT, (long)TT * HH, (long)TT * HH, 0);
  cvt(state_f, ctx_b, (long)BT * HH);

  // -- 6. combine ---------------------------------------------------------
  gemm(true, dec_b, HH, wch_b, HH, xp_f, HH, b_combine, BT, HH, HH, 1, 0, 0, 0, 0);
  gemm(true, ctx_b, HH, wcc_b, HH, xp_f, HH, nullptr, BT, HH, HH, 1, 0, 0, 0, 1);
  tanh_f32_bf16<<<dim3((BT * HH + 255) / 256), dim3(256), 0, stream>>>(
      xp_f, comb_b, (long)BT * HH);

  // -- 7. output projection (dominant GEMM, 134 GFLOP) --------------------
  gemm(true, comb_b, HH, wout_b, HH, logits, VV, b_out, BT, VV, HH, 1, 0, 0, 0, 0);
}
